// MaskedGraphAutoencoder_56659208568900
// MI455X (gfx1250) — matrix operand order repaired and checked by reading, not compiled, not run on hardware
//
#include <hip/hip_runtime.h>

#define NN      16384
#define EE      524288
#define ADJ_WAVES 4

typedef __attribute__((ext_vector_type(16))) __bf16 v16bf;
typedef __attribute__((ext_vector_type(8)))  __bf16 v8bf;
typedef __attribute__((ext_vector_type(8)))  float  v8f;

// -------- m = relu(x @ W^T + b),  W: [cout, cin] row-major ----------------
__global__ void fc_relu_kernel(const float* __restrict__ x,
                               const float* __restrict__ W,
                               const float* __restrict__ b,
                               float* __restrict__ y,
                               int n_rows, int cin, int cout) {
    int t = blockIdx.x * blockDim.x + threadIdx.x;
    if (t >= n_rows * cout) return;
    int c = t % cout, n = t / cout;
    const float* xr = x + (size_t)n * cin;
    const float* wr = W + (size_t)c * cin;
    float acc = b[c];
    for (int k = 0; k < cin; k += 4) {
        float4 xv = *(const float4*)(xr + k);
        float4 wv = *(const float4*)(wr + k);
        acc += xv.x * wv.x + xv.y * wv.y + xv.z * wv.z + xv.w * wv.w;
    }
    y[t] = fmaxf(acc, 0.0f);
}

__global__ void zero_kernel(float* __restrict__ p, int n) {
    int t = blockIdx.x * blockDim.x + threadIdx.x;
    if (t < n) p[t] = 0.0f;
}

// -------- neigh[dst] = segment_max(m[src] * ew)  (all values >= 0) --------
__global__ void scatter_max_kernel(const float* __restrict__ m,
                                   const int* __restrict__ src,
                                   const int* __restrict__ dst,
                                   const float* __restrict__ ew,
                                   float* __restrict__ neigh,
                                   int nedges, int cin) {
    int qn = cin >> 2;
    int t = blockIdx.x * blockDim.x + threadIdx.x;
    if (t >= nedges * qn) return;
    int e = t / qn;
    int q = (t - e * qn) << 2;
    int s = src[e], d = dst[e];
    float w = ew[e];
    float4 mv = *(const float4*)(m + (size_t)s * cin + q);
    unsigned* np = (unsigned*)(neigh + (size_t)d * cin + q);
    atomicMax(np + 0, __float_as_uint(mv.x * w));
    atomicMax(np + 1, __float_as_uint(mv.y * w));
    atomicMax(np + 2, __float_as_uint(mv.z * w));
    atomicMax(np + 3, __float_as_uint(mv.w * w));
}

// -------- y = relu(x @ Ws^T + neigh @ Wn^T + b) ---------------------------
__global__ void sage_out_kernel(const float* __restrict__ x,
                                const float* __restrict__ neigh,
                                const float* __restrict__ Ws,
                                const float* __restrict__ Wn,
                                const float* __restrict__ b,
                                float* __restrict__ y,
                                int n_rows, int cin, int cout) {
    int t = blockIdx.x * blockDim.x + threadIdx.x;
    if (t >= n_rows * cout) return;
    int c = t % cout, n = t / cout;
    const float* xr  = x + (size_t)n * cin;
    const float* nr  = neigh + (size_t)n * cin;
    const float* wsr = Ws + (size_t)c * cin;
    const float* wnr = Wn + (size_t)c * cin;
    float acc = b[c];
    for (int k = 0; k < cin; k += 4) {
        float4 xv = *(const float4*)(xr + k);
        float4 sv = *(const float4*)(wsr + k);
        float4 nv = *(const float4*)(nr + k);
        float4 wv = *(const float4*)(wnr + k);
        acc += xv.x * sv.x + xv.y * sv.y + xv.z * sv.z + xv.w * sv.w;
        acc += nv.x * wv.x + nv.y * wv.y + nv.z * wv.z + nv.w * wv.w;
    }
    y[t] = fmaxf(acc, 0.0f);
}

// -------- split fp32 -> bf16 hi + bf16 lo (RNE) ---------------------------
__global__ void split_bf16_kernel(const float* __restrict__ hd,
                                  unsigned short* __restrict__ hi,
                                  unsigned short* __restrict__ lo, int n) {
    int t = blockIdx.x * blockDim.x + threadIdx.x;
    if (t >= n) return;
    float f = hd[t];
    unsigned u  = __float_as_uint(f);
    unsigned rh = (u + 0x7FFFu + ((u >> 16) & 1u)) >> 16;
    hi[t] = (unsigned short)rh;
    float fl = f - __uint_as_float(rh << 16);
    unsigned ul = __float_as_uint(fl);
    unsigned rl = (ul + 0x7FFFu + ((ul >> 16) & 1u)) >> 16;
    lo[t] = (unsigned short)rl;
}

// -------- adj = hd @ hd^T via split-bf16 WMMA, K=64 -----------------------
// One wave computes one 16x16 f32 tile; 6x v_wmma_f32_16x16x32_bf16 per tile.
__global__ void adj_wmma_kernel(const __bf16* __restrict__ hi,
                                const __bf16* __restrict__ lo,
                                float* __restrict__ out) {
    const int lane = threadIdx.x & 31;
    const int wave = threadIdx.x >> 5;
    const int jt   = blockIdx.x * ADJ_WAVES + wave;
    const int it   = blockIdx.y;
    const int g    = lane >> 4;     // half-wave select
    const int r16  = lane & 15;

    const size_t rowA = (size_t)(it * 16 + r16) * 64;
    const size_t rowB = (size_t)(jt * 16 + r16) * 64;

    v8f acc = {};
#pragma unroll
    for (int kbase = 0; kbase < 64; kbase += 32) {
        // A 16x32 bf16: per-lane halves are K = kbase + g*16 + [0..15]
        const int ka = kbase + g * 16;
        v16bf Ah = *(const v16bf*)(hi + rowA + ka);
        v16bf Al = *(const v16bf*)(lo + rowA + ka);
        // B 32x16 bf16: per-lane halves are K = kbase + g*16 + [0..15]
        const int kb = kbase + g * 16;
        v16bf Bh = *(const v16bf*)(hi + rowB + kb);
        v16bf Bl = *(const v16bf*)(lo + rowB + kb);
        // x*y ~= hi*hi + hi*lo + lo*hi  (near-fp32 precision)
        acc = __builtin_amdgcn_wmma_f32_16x16x32_bf16(false, Ah, false, Bh, (short)0, acc, false, false);
        acc = __builtin_amdgcn_wmma_f32_16x16x32_bf16(false, Ah, false, Bl, (short)0, acc, false, false);
        acc = __builtin_amdgcn_wmma_f32_16x16x32_bf16(false, Al, false, Bh, (short)0, acc, false, false);
    }

    float* orow = out + (size_t)(it * 16 + g * 8) * NN + (size_t)jt * 16 + r16;
#pragma unroll
    for (int r = 0; r < 8; ++r) orow[(size_t)r * NN] = acc[r];
}

extern "C" void kernel_launch(void* const* d_in, const int* in_sizes, int n_in,
                              void* d_out, int out_size, void* d_ws, size_t ws_size,
                              hipStream_t stream) {
    const float* feat = (const float*)d_in[0];
    const int*   src  = (const int*)d_in[1];
    const int*   dst  = (const int*)d_in[2];
    const float* ew   = (const float*)d_in[3];

    const size_t BUF = (size_t)NN * 128 * sizeof(float);   // 8 MB
    char* ws = (char*)d_ws;
    float* bufA = (float*)(ws);
    float* bufB = (float*)(ws + BUF);
    float* mbuf = (float*)(ws + 2 * BUF);
    float* nbuf = (float*)(ws + 3 * BUF);
    // hi/lo reuse the (freed) message buffer after the layer phase
    unsigned short* hi = (unsigned short*)mbuf;
    unsigned short* lo = hi + (size_t)NN * 64;

    float* hd_out = (float*)d_out;                          // [N,64] final hidden
    float* adj    = hd_out + (size_t)NN * 64;               // [N,N]

    const int cins[6]  = {64, 64, 128, 32, 128, 128};
    const int couts[6] = {64, 128, 32, 128, 128, 64};
    float* houts[6]    = {bufA, bufB, bufA, bufB, bufA, hd_out};

    const float* hcur = feat;
    for (int l = 0; l < 6; ++l) {
        const int cin = cins[l], cout = couts[l];
        const float* wp = (const float*)d_in[4 + 5 * l + 0];
        const float* bp = (const float*)d_in[4 + 5 * l + 1];
        const float* Ws = (const float*)d_in[4 + 5 * l + 2];
        const float* Wn = (const float*)d_in[4 + 5 * l + 3];
        const float* bb = (const float*)d_in[4 + 5 * l + 4];

        int nt;
        nt = NN * cin;
        fc_relu_kernel<<<(nt + 255) / 256, 256, 0, stream>>>(hcur, wp, bp, mbuf, NN, cin, cin);
        zero_kernel<<<(nt + 255) / 256, 256, 0, stream>>>(nbuf, nt);
        nt = EE * (cin >> 2);
        scatter_max_kernel<<<(nt + 255) / 256, 256, 0, stream>>>(mbuf, src, dst, ew, nbuf, EE, cin);
        nt = NN * cout;
        sage_out_kernel<<<(nt + 255) / 256, 256, 0, stream>>>(hcur, nbuf, Ws, Wn, bb, houts[l], NN, cin, cout);
        hcur = houts[l];
    }

    // split hd into bf16 hi/lo for WMMA
    int nhd = NN * 64;
    split_bf16_kernel<<<(nhd + 255) / 256, 256, 0, stream>>>(hd_out, hi, lo, nhd);

    // adj = hd @ hd^T : exact tile coverage, 4 waves (= 4 tiles) per block
    dim3 gridAdj((NN / 16) / ADJ_WAVES, NN / 16);
    adj_wmma_kernel<<<gridAdj, ADJ_WAVES * 32, 0, stream>>>((const __bf16*)hi, (const __bf16*)lo, adj);
}
